// DilatedSelfAttention_57621281243334
// MI455X (gfx1250) — matrix operand order repaired and checked by reading, not compiled, not run on hardware
//
#include <hip/hip_runtime.h>
#include <hip/hip_bf16.h>
#include <stdint.h>

// MI455X / gfx1250: wave32, WMMA bf16 16x16x32, f32 accumulate.
typedef __bf16 bf16;
typedef __attribute__((ext_vector_type(16))) __bf16 v16bf;
typedef __attribute__((ext_vector_type(8)))  __bf16 v8bf;
typedef __attribute__((ext_vector_type(8)))  float  v8f;

#define B_   4
#define N_   8192
#define C_   256
#define D_   256
#define NTOK (B_ * N_)

#if defined(__has_builtin)
#if __has_builtin(__builtin_amdgcn_global_load_async_to_lds_b128) && \
    __has_builtin(__builtin_amdgcn_s_wait_asynccnt)
#define HAVE_ASYNC_LDS 1
#endif
#endif
#ifndef HAVE_ASYNC_LDS
#define HAVE_ASYNC_LDS 0
#endif

#if HAVE_ASYNC_LDS
// Builtin expects: (v4i global*, v4i shared*, imm offset, imm cpol)
typedef int v4i_vs __attribute__((vector_size(16)));
typedef __attribute__((address_space(1))) v4i_vs* g4p;
typedef __attribute__((address_space(3))) v4i_vs* l4p;
#endif

__device__ __forceinline__ v16bf cat8(v8bf a, v8bf b) {
  return __builtin_shufflevector(a, b, 0, 1, 2, 3, 4, 5, 6, 7,
                                       8, 9, 10, 11, 12, 13, 14, 15);
}

__device__ __forceinline__ v16bf ldfrag(const bf16* p) {
  return cat8(*(const v8bf*)p, *(const v8bf*)(p + 8));
}

__device__ __forceinline__ uint32_t pack_bf16(float lo, float hi) {
  union { __bf16 h[2]; uint32_t u; } t;
  t.h[0] = (__bf16)lo;
  t.h[1] = (__bf16)hi;
  return t.u;
}

__device__ __forceinline__ v8f wmma_bf16(v16bf a, v16bf b, v8f c) {
  return __builtin_amdgcn_wmma_f32_16x16x32_bf16(
      false, a, false, b, (short)0, c, false, false);
}

// ---------------------------------------------------------------------------
// Q/K/V projection: [32768,256] x [256,256] -> bf16 (all three weight mats).
// grid = (512, 3), block = 256 (8 waves). WG tile: 64 rows x 256 cols.
__global__ void __launch_bounds__(256) dsa_proj_kernel(
    const float* __restrict__ x,
    const float* __restrict__ Wq, const float* __restrict__ Wk,
    const float* __restrict__ Wv,
    bf16* __restrict__ Qb, bf16* __restrict__ Kb, bf16* __restrict__ Vb) {
  __shared__ __align__(16) bf16 sX[64][48];    // [row][k]
  __shared__ __align__(16) bf16 sWt[256][48];  // [n][k] (transposed)

  const int tid  = threadIdx.x;
  const int lane = tid & 31;
  const int wv   = tid >> 5;
  const int rowgroup = wv >> 1;
  const int colhalf  = wv & 1;
  const int mat      = blockIdx.y;
  const float* W = (mat == 0) ? Wq : (mat == 1) ? Wk : Wv;
  bf16*      dst = (mat == 0) ? Qb : (mat == 1) ? Kb : Vb;
  const int rowblock = blockIdx.x * 64;
  const int half8  = (lane >= 16) ? 8 : 0;
  const int half16 = (lane >= 16) ? 16 : 0;
  const int l15    = lane & 15;

  const v8f z8 = {0.f, 0.f, 0.f, 0.f, 0.f, 0.f, 0.f, 0.f};
  v8f acc[8];
#pragma unroll
  for (int i = 0; i < 8; ++i) acc[i] = z8;

  for (int ks = 0; ks < 8; ++ks) {
    // ---- stage x tile (64 x 32), f32 pairs -> packed bf16 dwords
#pragma unroll
    for (int i = 0; i < 4; ++i) {
      int e2 = tid + i * 256;
      int row = e2 >> 4, kp = e2 & 15;
      float2 xv = ((const float2*)x)[(size_t)(rowblock + row) * 128 + ks * 16 + kp];
      *(uint32_t*)&sX[row][kp * 2] = pack_bf16(xv.x, xv.y);
    }
    // ---- stage W tile transposed (32k x 256n -> sWt[n][k]), 4 k packed/b64
#pragma unroll
    for (int i = 0; i < 2; ++i) {
      int g2 = tid + i * 256;
      int kq = g2 >> 6, nq = g2 & 63;
      int kb0 = ks * 32 + kq * 4;
      float4 r0 = ((const float4*)W)[(size_t)(kb0 + 0) * 64 + nq];
      float4 r1 = ((const float4*)W)[(size_t)(kb0 + 1) * 64 + nq];
      float4 r2 = ((const float4*)W)[(size_t)(kb0 + 2) * 64 + nq];
      float4 r3 = ((const float4*)W)[(size_t)(kb0 + 3) * 64 + nq];
      const float* f0 = &r0.x; const float* f1 = &r1.x;
      const float* f2 = &r2.x; const float* f3 = &r3.x;
#pragma unroll
      for (int n = 0; n < 4; ++n) {
        uint2 pk;
        pk.x = pack_bf16(f0[n], f1[n]);
        pk.y = pack_bf16(f2[n], f3[n]);
        *(uint2*)&sWt[nq * 4 + n][kq * 4] = pk;
      }
    }
    __syncthreads();

    const int arow = rowgroup * 16 + l15;
    v16bf a = cat8(*(const v8bf*)&sX[arow][half16],
                   *(const v8bf*)&sX[arow][half16 + 8]);
    // 2-deep pipelined B fragments (two live buffers, no coalescing)
    v16bf b0 = ldfrag(&sWt[colhalf * 128 + 0 * 16 + l15][half16]);
    v16bf b1 = ldfrag(&sWt[colhalf * 128 + 1 * 16 + l15][half16]);
#pragma unroll
    for (int nt = 0; nt < 8; nt += 2) {
      acc[nt] = wmma_bf16(a, b0, acc[nt]);
      if (nt + 2 < 8)
        b0 = ldfrag(&sWt[colhalf * 128 + (nt + 2) * 16 + l15][half16]);
      acc[nt + 1] = wmma_bf16(a, b1, acc[nt + 1]);
      if (nt + 3 < 8)
        b1 = ldfrag(&sWt[colhalf * 128 + (nt + 3) * 16 + l15][half16]);
    }
    __syncthreads();
  }

#pragma unroll
  for (int nt = 0; nt < 8; ++nt) {
#pragma unroll
    for (int v = 0; v < 8; ++v) {
      int row = rowblock + rowgroup * 16 + v + ((lane >= 16) ? 8 : 0);
      int col = colhalf * 128 + nt * 16 + l15;
      dst[(size_t)row * D_ + col] = (bf16)acc[nt][v];
    }
  }
}

// ---------------------------------------------------------------------------
// Flash causal attention for one dilation config. grid = (16, segs*B),
// block = 256 (8 waves). Wave: 16 q rows x full D=256. K/V double-buffered
// in LDS: async-stage block kb+1 while computing on block kb, one barrier
// per key block. accumulate==0: plain stores (config 0 covers all tokens);
// accumulate==1: single-touch atomic adds (kernels stream-serialized).
__global__ void __launch_bounds__(256) dsa_attn_kernel(
    const bf16* __restrict__ Qb, const bf16* __restrict__ Kb,
    const bf16* __restrict__ Vb,
    float* __restrict__ out, float* __restrict__ den,
    int w, int r, int accumulate) {
  __shared__ __align__(16) bf16 sK[2][32][264];    // [buf][key][d]
  __shared__ __align__(16) bf16 sVt[2][256][48];   // [buf][d][key]
  __shared__ __align__(16) bf16 sP[8][16][48];     // per-wave P tile [q][key]

  const int tid   = threadIdx.x;
  const int lane  = tid & 31;
  const int wv    = tid >> 5;
  const int sb    = blockIdx.y;
  const int seg   = sb >> 2;  // B_ == 4
  const int b     = sb & 3;
  const int qbase = blockIdx.x * 128;
  const int segoff = b * N_ + seg * w;
  const int half8  = (lane >= 16) ? 8 : 0;
  const int half16 = (lane >= 16) ? 16 : 0;
  const int l15    = lane & 15;

  // Cooperative K/V staging of one 32-key block into buffer bsel.
  auto stage = [&](int kbase, int bsel) {
    // K: 32 rows x 128 dwords (async global->LDS when available)
#pragma unroll
    for (int i = 0; i < 4; ++i) {
      int u = tid + i * 256;
      int row = u >> 5, q = u & 31;
      size_t tok = (size_t)(segoff + (kbase + row) * r);
#if HAVE_ASYNC_LDS
      __builtin_amdgcn_global_load_async_to_lds_b128(
          (g4p)((const uint4*)Kb + tok * 32 + q),
          (l4p)&sK[bsel][row][q * 8], 0, 0);
#else
      *(uint4*)&sK[bsel][row][q * 8] = ((const uint4*)Kb)[tok * 32 + q];
#endif
    }
    // V transposed: keys 4kg..4kg+3 x d-quad qd, 4 keys packed per b64
    {
      int kg = tid >> 5;
      int qd = tid & 31;
      size_t t0 = (size_t)(segoff + (kbase + 4 * kg + 0) * r);
      size_t t1 = (size_t)(segoff + (kbase + 4 * kg + 1) * r);
      size_t t2 = (size_t)(segoff + (kbase + 4 * kg + 2) * r);
      size_t t3 = (size_t)(segoff + (kbase + 4 * kg + 3) * r);
      uint4 A0 = ((const uint4*)Vb)[t0 * 32 + qd];
      uint4 A1 = ((const uint4*)Vb)[t1 * 32 + qd];
      uint4 A2 = ((const uint4*)Vb)[t2 * 32 + qd];
      uint4 A3 = ((const uint4*)Vb)[t3 * 32 + qd];
      const uint32_t* w0 = &A0.x; const uint32_t* w1 = &A1.x;
      const uint32_t* w2 = &A2.x; const uint32_t* w3 = &A3.x;
#pragma unroll
      for (int q = 0; q < 4; ++q) {
        int d0 = qd * 8 + 2 * q;
        uint2 lo, hi;
        lo.x = (w0[q] & 0xffffu) | (w1[q] << 16);
        lo.y = (w2[q] & 0xffffu) | (w3[q] << 16);
        hi.x = (w0[q] >> 16)     | (w1[q] & 0xffff0000u);
        hi.y = (w2[q] >> 16)     | (w3[q] & 0xffff0000u);
        *(uint2*)&sVt[bsel][d0][4 * kg]     = lo;
        *(uint2*)&sVt[bsel][d0 + 1][4 * kg] = hi;
      }
    }
  };

  // Q A-fragments (16 rows x 256 d) in VGPRs, vector-loaded from global.
  v16bf aq[8];
  {
    const int qm = qbase + wv * 16 + l15;
    const size_t gq = (size_t)(segoff + qm * r) * D_;
#pragma unroll
    for (int d = 0; d < 8; ++d) {
      aq[d] = cat8(*(const v8bf*)(Qb + gq + d * 32 + half16),
                   *(const v8bf*)(Qb + gq + d * 32 + half16 + 8));
    }
  }

  const v8f z8 = {0.f, 0.f, 0.f, 0.f, 0.f, 0.f, 0.f, 0.f};
  v8f acc[16];
#pragma unroll
  for (int i = 0; i < 16; ++i) acc[i] = z8;
  float mrun[8], lsum[8];
#pragma unroll
  for (int v = 0; v < 8; ++v) { mrun[v] = -1e30f; lsum[v] = 0.0f; }

  const int qlast = qbase + wv * 16 + 15;
  const int nkb   = (qbase + 128) >> 5;

  stage(0, 0);
#if HAVE_ASYNC_LDS
  __builtin_amdgcn_s_wait_asynccnt(0);
#endif
  __syncthreads();

  for (int kb = 0; kb < nkb; ++kb) {
    const int kbase = kb * 32;
    const int cur = kb & 1;

    // Stage next block into the other buffer while computing this one.
    if (kb + 1 < nkb) stage(kbase + 32, cur ^ 1);
    if (kb + 2 < nkb) {  // prefetch block kb+2 (global_prefetch_b8)
      int prow = tid & 31, pch = (tid >> 5) & 3;
      size_t ptok = (size_t)(segoff + (kbase + 64 + prow) * r);
      __builtin_prefetch(Kb + ptok * D_ + pch * 64, 0, 1);
      __builtin_prefetch(Vb + ptok * D_ + pch * 64, 0, 1);
    }

    if (kbase <= qlast) {  // causal early-out per wave
      // ---- S = Q K^T, 2-deep pipelined fragment loads
      v8f s0 = z8, s1 = z8;
      v16bf bk0 = ldfrag(&sK[cur][l15][half16]);
      v16bf bk1 = ldfrag(&sK[cur][16 + l15][half16]);
#pragma unroll
      for (int d = 0; d < 8; ++d) {
        s0 = wmma_bf16(aq[d], bk0, s0);
        if (d + 1 < 8) bk0 = ldfrag(&sK[cur][l15][(d + 1) * 32 + half16]);
        s1 = wmma_bf16(aq[d], bk1, s1);
        if (d + 1 < 8) bk1 = ldfrag(&sK[cur][16 + l15][(d + 1) * 32 + half16]);
      }
      // ---- online softmax (row = v + half8, reduced over 16-lane half)
      float alpha[8];
#pragma unroll
      for (int v = 0; v < 8; ++v) {
        float e0 = s0[v] * 0.0625f;  // 1/sqrt(256)
        float e1 = s1[v] * 0.0625f;
        const int qm = qbase + wv * 16 + v + half8;
        const int k0 = kbase + l15;
        if (k0 > qm)      e0 = -1e30f;
        if (k0 + 16 > qm) e1 = -1e30f;
        float rm = fmaxf(e0, e1);
        rm = fmaxf(rm, __shfl_xor(rm, 1));
        rm = fmaxf(rm, __shfl_xor(rm, 2));
        rm = fmaxf(rm, __shfl_xor(rm, 4));
        rm = fmaxf(rm, __shfl_xor(rm, 8));
        const float mn = fmaxf(mrun[v], rm);
        const float al = __expf(mrun[v] - mn);
        const float p0 = __expf(e0 - mn);
        const float p1 = __expf(e1 - mn);
        float rs = p0 + p1;
        rs += __shfl_xor(rs, 1);
        rs += __shfl_xor(rs, 2);
        rs += __shfl_xor(rs, 4);
        rs += __shfl_xor(rs, 8);
        lsum[v] = lsum[v] * al + rs;
        mrun[v] = mn;
        alpha[v] = al;
        const int prow = v + half8;
        sP[wv][prow][l15]      = (bf16)p0;
        sP[wv][prow][l15 + 16] = (bf16)p1;
      }
#pragma unroll
      for (int nt = 0; nt < 16; ++nt) {
#pragma unroll
        for (int v = 0; v < 8; ++v) acc[nt][v] *= alpha[v];
      }
      // ---- O += P * V, 2-deep pipelined B fragments from transposed V
      v16bf ap = cat8(*(const v8bf*)&sP[wv][l15][half16],
                      *(const v8bf*)&sP[wv][l15][half16 + 8]);
      v16bf b0 = ldfrag(&sVt[cur][0 * 16 + l15][half16]);
      v16bf b1 = ldfrag(&sVt[cur][1 * 16 + l15][half16]);
#pragma unroll
      for (int nt = 0; nt < 16; nt += 2) {
        acc[nt] = wmma_bf16(ap, b0, acc[nt]);
        if (nt + 2 < 16) b0 = ldfrag(&sVt[cur][(nt + 2) * 16 + l15][half16]);
        acc[nt + 1] = wmma_bf16(ap, b1, acc[nt + 1]);
        if (nt + 3 < 16) b1 = ldfrag(&sVt[cur][(nt + 3) * 16 + l15][half16]);
      }
    }
#if HAVE_ASYNC_LDS
    __builtin_amdgcn_s_wait_asynccnt(0);
#endif
    __syncthreads();
  }

  // numerator / denominator write-out.
#pragma unroll
  for (int v = 0; v < 8; ++v) {
    const float es = __expf(mrun[v]);
    const int qm = qbase + wv * 16 + v + half8;
    const size_t g = (size_t)(segoff + qm * r);
    if (accumulate) {
#pragma unroll
      for (int nt = 0; nt < 16; ++nt)
        atomicAdd(&out[g * D_ + nt * 16 + l15], acc[nt][v] * es);
      if (l15 == 0) atomicAdd(&den[g], lsum[v] * es);
    } else {
#pragma unroll
      for (int nt = 0; nt < 16; ++nt)
        out[g * D_ + nt * 16 + l15] = acc[nt][v] * es;
      if (l15 == 0) den[g] = lsum[v] * es;
    }
  }
}

// ---------------------------------------------------------------------------
__global__ void dsa_norm_kernel(float4* __restrict__ out4,
                                const float* __restrict__ den) {
  int i = blockIdx.x * blockDim.x + threadIdx.x;
  if (i < NTOK * D_ / 4) {
    const float inv = 1.0f / den[i >> 6];
    float4 v = out4[i];
    v.x *= inv; v.y *= inv; v.z *= inv; v.w *= inv;
    out4[i] = v;
  }
}

// ---------------------------------------------------------------------------
extern "C" void kernel_launch(void* const* d_in, const int* in_sizes, int n_in,
                              void* d_out, int out_size, void* d_ws,
                              size_t ws_size, hipStream_t stream) {
  (void)in_sizes; (void)n_in; (void)out_size; (void)ws_size;
  const float* x  = (const float*)d_in[0];
  const float* Wq = (const float*)d_in[1];
  const float* Wk = (const float*)d_in[2];
  const float* Wv = (const float*)d_in[3];
  float* out = (float*)d_out;

  // Workspace (~48.1 MB): Qb/Kb/Vb bf16 [32768,256]; den f32 [32768].
  bf16* Qb = (bf16*)d_ws;
  bf16* Kb = Qb + (size_t)NTOK * D_;
  bf16* Vb = Kb + (size_t)NTOK * D_;
  float* den = (float*)(Vb + (size_t)NTOK * D_);

  dsa_proj_kernel<<<dim3(NTOK / 64, 3), 256, 0, stream>>>(
      x, Wq, Wk, Wv, Qb, Kb, Vb);

  // Config 0 covers every token: plain stores initialize out/den.
  dsa_attn_kernel<<<dim3(16, 4 * B_), 256, 0, stream>>>(
      Qb, Kb, Vb, out, den, 2048, 1, 0);
  // Configs 1/2: single-touch atomic accumulation (stream-serialized).
  dsa_attn_kernel<<<dim3(16, 2 * B_), 256, 0, stream>>>(
      Qb, Kb, Vb, out, den, 4096, 2, 1);
  dsa_attn_kernel<<<dim3(16, 1 * B_), 256, 0, stream>>>(
      Qb, Kb, Vb, out, den, 8192, 4, 1);

  dsa_norm_kernel<<<(NTOK * D_ / 4 + 255) / 256, 256, 0, stream>>>(
      (float4*)out, den);
}
